// VQVAETwitterizerOC_14585708937920
// MI455X (gfx1250) — matrix operand order repaired and checked by reading, not compiled, not run on hardware
//
#include <hip/hip_runtime.h>
#include <math.h>

typedef __attribute__((ext_vector_type(16))) _Float16 v16h;
typedef __attribute__((ext_vector_type(8)))  _Float16 h8;
typedef __attribute__((ext_vector_type(8)))  float    v8f;

#define NB    32
#define TFULL 8192
#define TP    1024
#define HC    128
#define EMB   64
#define KC    512
#define PATCH 8

// ---------------------------------------------------------------------------
// WMMA fragment helpers (layouts per CDNA5 ISA 7.12.2, wave32)
// A (16x32 f16, row-major source, row stride ld):
//   lane L: row = rowbase + (L&15); K chunk base cb = (L&16)?8:0
//   halves [0..7] = K kb+cb+0..7, halves [8..15] = K kb+cb+16..23
__device__ __forceinline__ v16h frag_a_rm(const _Float16* M, int ld, int rowbase,
                                          int kb, int lane) {
  const _Float16* p = M + (size_t)(rowbase + (lane & 15)) * ld + kb + ((lane & 16) ? 8 : 0);
  h8 lo = *(const h8*)p;
  h8 hi = *(const h8*)(p + 16);
  return __builtin_shufflevector(lo, hi, 0,1,2,3,4,5,6,7,8,9,10,11,12,13,14,15);
}
// B (32x16 f16) from a K-major buffer: column stride ld, K contiguous.
//   lane L: col = colbase + (L&15); cb = (L&16)?8:0; halves [0..7] = K kb+cb+0..7, [8..15] = K kb+cb+16..23
__device__ __forceinline__ v16h frag_b_km(const _Float16* P, int ld, int colbase,
                                          int kb, int lane) {
  const _Float16* p = P + (size_t)(colbase + (lane & 15)) * ld + kb + ((lane & 16) ? 8 : 0);
  h8 lo = *(const h8*)p;
  h8 hi = *(const h8*)(p + 16);
  return __builtin_shufflevector(lo, hi, 0,1,2,3,4,5,6,7,8,9,10,11,12,13,14,15);
}
#define WMMA_F16(a, b, c) \
  __builtin_amdgcn_wmma_f32_16x16x32_f16(false, (a), false, (b), (short)0, (c), false, false)

// ---------------------------------------------------------------------------
// Prep: convert weights to f16 arenas, code norms, zero accumulators
__device__ __forceinline__ void cvt_arr(const float* s, _Float16* d, int n, int tid, int nth) {
  for (int i = tid; i < n; i += nth) d[i] = (_Float16)s[i];
}

__global__ void k_prep(const float* ew2, const float* er0a, const float* er0b,
                       const float* er1a, const float* er1b, const float* ewo,
                       const float* cb, const float* dw1,
                       const float* dr0a, const float* dr0b,
                       const float* dr1a, const float* dr1b, const float* dwt,
                       _Float16* W2E, _Float16* R0A, _Float16* R0B, _Float16* R1A,
                       _Float16* R1B, _Float16* WOE, _Float16* CBH, float* CSQ,
                       _Float16* DW1o, _Float16* DR0Ao, _Float16* DR0Bo,
                       _Float16* DR1Ao, _Float16* DR1Bo, _Float16* WTT, float* scal) {
  int tid = blockIdx.x * blockDim.x + threadIdx.x;
  int nth = gridDim.x * blockDim.x;
  cvt_arr(ew2,  W2E, HC * HC * PATCH, tid, nth);       // K = c*8+k, matches (O,I,k)
  cvt_arr(er0a, R0A, HC * HC * 3, tid, nth);
  cvt_arr(er0b, R0B, HC * HC, tid, nth);
  cvt_arr(er1a, R1A, HC * HC * 3, tid, nth);
  cvt_arr(er1b, R1B, HC * HC, tid, nth);
  cvt_arr(ewo,  WOE, EMB * HC, tid, nth);
  cvt_arr(cb,   CBH, KC * EMB, tid, nth);
  cvt_arr(dw1,  DW1o, HC * EMB * 3, tid, nth);
  cvt_arr(dr0a, DR0Ao, HC * HC * 3, tid, nth);
  cvt_arr(dr0b, DR0Bo, HC * HC, tid, nth);
  cvt_arr(dr1a, DR1Ao, HC * HC * 3, tid, nth);
  cvt_arr(dr1b, DR1Bo, HC * HC, tid, nth);
  // dec_wt (I=128,O=128,k=8) -> WTT[m=c'*8+k][c]
  for (int i = tid; i < HC * HC * PATCH; i += nth) {
    int c = i >> 10;           // input channel
    int rest = i & 1023;       // c'*8 + k
    WTT[(size_t)rest * HC + c] = (_Float16)dwt[i];
  }
  for (int k = tid; k < KC; k += nth) {
    float s = 0.f;
    for (int d = 0; d < EMB; d++) { float v = cb[k * EMB + d]; s += v * v; }
    CSQ[k] = s;
  }
  for (int i = tid; i < KC + 2; i += nth) scal[i] = 0.f;
}

// ---------------------------------------------------------------------------
// Fused encoder front: conv1(k3,p1)+ReLU computed in LDS, then stride-8 conv2
// as GEMM  W2[128][1024] x patch[1024][16]  -> ACT[b][128][t'16-tile]
__global__ void __launch_bounds__(256) k_enc_front(const float* __restrict__ x,
                                                   const float* __restrict__ w1,
                                                   const float* __restrict__ b1,
                                                   const _Float16* __restrict__ W2E,
                                                   const float* __restrict__ b2,
                                                   _Float16* __restrict__ out) {
  __shared__ float xs[132];
  __shared__ _Float16 P[16 * 1032];   // [n][K=c*8+k], ld padded to 1032
  const int tile = blockIdx.x;        // 64 tiles of 16 t'
  const int b = blockIdx.y;
  const int tid = threadIdx.x;
  const int t0 = tile * 16;

  for (int i = tid; i < 130; i += 256) {
    int gx = t0 * PATCH - 1 + i;
    xs[i] = (gx >= 0 && gx < TFULL) ? x[(size_t)b * TFULL + gx] : 0.f;
  }
  __syncthreads();
  // h[c][tt] = relu(w1[c]·x + b1[c]), tt in [0,128)
  for (int idx = tid; idx < HC * 128; idx += 256) {
    int c = idx >> 7, tt = idx & 127;
    float v = fmaf(w1[c * 3 + 0], xs[tt],
              fmaf(w1[c * 3 + 1], xs[tt + 1],
              fmaf(w1[c * 3 + 2], xs[tt + 2], b1[c])));
    v = fmaxf(v, 0.f);
    P[(tt >> 3) * 1032 + c * PATCH + (tt & 7)] = (_Float16)v;
  }
  __syncthreads();

  const int w = tid >> 5, lane = tid & 31;
  const int mb = w * 16;
  v8f acc = {};
  __builtin_prefetch(W2E + (size_t)mb * 1024, 0, 1);
  for (int kb = 0; kb < 1024; kb += 32) {
    v16h a = frag_a_rm(W2E, 1024, mb, kb, lane);
    v16h bf = frag_b_km(P, 1032, 0, kb, lane);
    acc = WMMA_F16(a, bf, acc);
  }
  const int N = t0 + (lane & 15);
#pragma unroll
  for (int r = 0; r < 8; r++) {
    int M = mb + r + ((lane & 16) ? 8 : 0);
    float v = acc[r] + b2[M];
    out[((size_t)b * HC + M) * TP + N] = (_Float16)v;
  }
}

// ---------------------------------------------------------------------------
// Generic conv-as-GEMM over T'=1024: out[b][o][t] = bias + W[o][c*KT+k]*in[c][t+k-pad]
__global__ void __launch_bounds__(256) k_conv_gemm(const _Float16* __restrict__ in,
                                                   const _Float16* __restrict__ W,
                                                   const float* __restrict__ bias,
                                                   const _Float16* __restrict__ residual,
                                                   void* __restrict__ out,
                                                   int I, int KT, int pad, int O,
                                                   int relu_in, int out_f32) {
  __shared__ _Float16 P[16 * 392];    // max: K=384 -> ld 392
  const int tile = blockIdx.x, b = blockIdx.y, tid = threadIdx.x;
  const int t0 = tile * 16;
  const int K = I * KT, ldP = K + 8;

  for (int idx = tid; idx < 16 * K; idx += 256) {
    int n = idx / K, kk = idx - n * K;
    int c = kk / KT, k = kk - c * KT;
    int t = t0 + n + k - pad;
    float v = 0.f;
    if (t >= 0 && t < TP) v = (float)in[((size_t)b * I + c) * TP + t];
    if (relu_in) v = fmaxf(v, 0.f);
    P[n * ldP + kk] = (_Float16)v;
  }
  __syncthreads();

  const int w = tid >> 5, lane = tid & 31;
  if (w < (O >> 4)) {
    const int mb = w * 16;
    v8f acc = {};
    __builtin_prefetch(W + (size_t)mb * K, 0, 1);
    for (int kb = 0; kb < K; kb += 32) {
      v16h a = frag_a_rm(W, K, mb, kb, lane);
      v16h bf = frag_b_km(P, ldP, 0, kb, lane);
      acc = WMMA_F16(a, bf, acc);
    }
    const int N = t0 + (lane & 15);
#pragma unroll
    for (int r = 0; r < 8; r++) {
      int M = mb + r + ((lane & 16) ? 8 : 0);
      float v = acc[r] + bias[M];
      size_t oi = ((size_t)b * O + M) * TP + N;
      if (residual) v += (float)residual[oi];
      if (out_f32) ((float*)out)[oi] = v;
      else         ((_Float16*)out)[oi] = (_Float16)v;
    }
  }
}

// ---------------------------------------------------------------------------
// Quantize: dists for 16 positions x 512 codes via WMMA (-2 z·c + |z|^2 + |c|^2),
// argmin, token ids, zq gather (f16 for decoder), commit-loss + histogram.
__global__ void __launch_bounds__(256) k_quantize(const float* __restrict__ ZE,
                                                  const _Float16* __restrict__ CBH,
                                                  const float* __restrict__ CSQ,
                                                  const float* __restrict__ cb32,
                                                  _Float16* __restrict__ ZQ,
                                                  float* __restrict__ dout,
                                                  float* __restrict__ scal) {
  __shared__ _Float16 zeh[16 * 72];
  __shared__ float zef[16 * 64];
  __shared__ float zesq[16];
  __shared__ float dist[16 * 512];
  const int tid = threadIdx.x;
  const int p0 = blockIdx.x * 16;
  const int b = p0 >> 10;

  for (int idx = tid; idx < 16 * EMB; idx += 256) {
    int r = idx >> 6, d = idx & 63;
    int tP = (p0 + r) & (TP - 1);
    float v = ZE[((size_t)b * EMB + d) * TP + tP];
    zef[r * 64 + d] = v;
    zeh[r * 72 + d] = (_Float16)v;
  }
  __syncthreads();
  if (tid < 16) {
    float s = 0.f;
    for (int d = 0; d < EMB; d++) { float v = zef[tid * 64 + d]; s += v * v; }
    zesq[tid] = s;
  }
  __syncthreads();

  const int w = tid >> 5, lane = tid & 31;
  for (int j = 0; j < 4; j++) {
    const int ct = w * 4 + j;           // 32 code tiles of 16
    v8f acc = {};
    for (int kb = 0; kb < EMB; kb += 32) {
      v16h a = frag_a_rm(zeh, 72, 0, kb, lane);
      v16h bf = frag_b_km(CBH, EMB, ct * 16, kb, lane);
      acc = WMMA_F16(a, bf, acc);
    }
    const int N = ct * 16 + (lane & 15);
    const float cq = CSQ[N];
#pragma unroll
    for (int r = 0; r < 8; r++) {
      int M = r + ((lane & 16) ? 8 : 0);
      dist[M * 512 + N] = zesq[M] + cq - 2.f * acc[r];
    }
  }
  __syncthreads();

  if (tid < 16) {
    const int r = tid;
    float best = dist[r * 512];
    int bk = 0;
    for (int k = 1; k < KC; k++) {
      float d = dist[r * 512 + k];
      if (d < best) { best = d; bk = k; }
    }
    const int p = p0 + r;
    const int tP = p & (TP - 1);
    dout[NB * TFULL + p] = (float)bk;               // token ids
    float cacc = 0.f;
    for (int d = 0; d < EMB; d++) {
      float cv = cb32[bk * EMB + d];
      ZQ[((size_t)b * EMB + d) * TP + tP] = (_Float16)cv;
      float df = zef[r * 64 + d] - cv;
      cacc += df * df;
    }
    atomicAdd(&scal[513], cacc);
    atomicAdd(&scal[bk], 1.0f);
  }
}

// ---------------------------------------------------------------------------
// Transposed conv k=stride=8 as GEMM: M=1024 (c'*8+k), K=128, N=16 t' -> relu
__global__ void __launch_bounds__(256) k_deconv(const _Float16* __restrict__ in,
                                                const _Float16* __restrict__ WTT,
                                                const float* __restrict__ bt,
                                                _Float16* __restrict__ hbig) {
  __shared__ _Float16 P[16 * 136];    // [n][c], ld 136
  const int tile = blockIdx.x, b = blockIdx.y, tid = threadIdx.x;
  const int t0 = tile * 16;

  for (int idx = tid; idx < 16 * HC; idx += 256) {
    int n = idx >> 7, c = idx & 127;
    float v = (float)in[((size_t)b * HC + c) * TP + t0 + n];
    P[n * 136 + c] = (_Float16)fmaxf(v, 0.f);       // relu before deconv
  }
  __syncthreads();

  const int w = tid >> 5, lane = tid & 31;
  v8f zero = {};
  v8f acc[8];
#pragma unroll
  for (int i = 0; i < 8; i++) acc[i] = zero;
  for (int kb = 0; kb < HC; kb += 32) {
    v16h bf = frag_b_km(P, 136, 0, kb, lane);
#pragma unroll
    for (int i = 0; i < 8; i++) {
      v16h a = frag_a_rm(WTT, HC, (w * 8 + i) * 16, kb, lane);
      acc[i] = WMMA_F16(a, bf, acc[i]);
    }
  }
  const int ncol = t0 + (lane & 15);
#pragma unroll
  for (int i = 0; i < 8; i++) {
#pragma unroll
    for (int r = 0; r < 8; r++) {
      int m = (w * 8 + i) * 16 + r + ((lane & 16) ? 8 : 0);
      int cp = m >> 3, k = m & 7;
      float v = fmaxf(acc[i][r] + bt[cp], 0.f);
      hbig[((size_t)b * HC + cp) * TFULL + ncol * PATCH + k] = (_Float16)v;
    }
  }
}

// ---------------------------------------------------------------------------
// Final O=1 conv (k3,p1) + x_hat store + recon-loss accumulation
__global__ void __launch_bounds__(256) k_final(const _Float16* __restrict__ hb,
                                               const float* __restrict__ wo,
                                               const float* __restrict__ bo,
                                               const float* __restrict__ x,
                                               float* __restrict__ dout,
                                               float* __restrict__ scal) {
  __shared__ float red[256];
  const int idx = blockIdx.x * 256 + threadIdx.x;
  const int b = idx >> 13, t = idx & (TFULL - 1);
  float s = bo[0];
  for (int c = 0; c < HC; c++) {
    const _Float16* row = hb + ((size_t)b * HC + c) * TFULL;
    const float* wr = wo + c * 3;
#pragma unroll
    for (int k = 0; k < 3; k++) {
      int tt = t + k - 1;
      if (tt >= 0 && tt < TFULL) s = fmaf((float)row[tt], wr[k], s);
    }
  }
  dout[idx] = s;
  float d = s - x[idx];
  red[threadIdx.x] = d * d;
  __syncthreads();
  for (int st = 128; st > 0; st >>= 1) {
    if (threadIdx.x < st) red[threadIdx.x] += red[threadIdx.x + st];
    __syncthreads();
  }
  if (threadIdx.x == 0) atomicAdd(&scal[512], red[0]);
}

// ---------------------------------------------------------------------------
// Losses + perplexity
__global__ void k_finalize(const float* __restrict__ scal, float* __restrict__ dout) {
  __shared__ float red[256];
  const int tid = threadIdx.x;
  float s = 0.f;
  for (int k = tid; k < KC; k += 256) {
    float avg = scal[k] / (float)(NB * TP);
    s += avg * logf(avg + 1e-10f);
  }
  red[tid] = s;
  __syncthreads();
  for (int st = 128; st > 0; st >>= 1) {
    if (tid < st) red[tid] += red[tid + st];
    __syncthreads();
  }
  if (tid == 0) {
    float recon = scal[512] / (float)(NB * TFULL);
    float commit = 0.25f * scal[513] / (float)(NB * EMB * TP);
    size_t base = (size_t)NB * TFULL + (size_t)NB * TP;
    dout[base + 0] = recon + commit;
    dout[base + 1] = recon;
    dout[base + 2] = commit;
    dout[base + 3] = expf(-red[0]);
  }
}

// ---------------------------------------------------------------------------
extern "C" void kernel_launch(void* const* d_in, const int* in_sizes, int n_in,
                              void* d_out, int out_size, void* d_ws, size_t ws_size,
                              hipStream_t stream) {
  (void)in_sizes; (void)n_in; (void)out_size; (void)ws_size;
  const float* x       = (const float*)d_in[0];
  const float* enc_w1  = (const float*)d_in[1];
  const float* enc_b1  = (const float*)d_in[2];
  const float* enc_w2  = (const float*)d_in[3];
  const float* enc_b2  = (const float*)d_in[4];
  const float* er0_wa  = (const float*)d_in[5];
  const float* er0_ba  = (const float*)d_in[6];
  const float* er0_wb  = (const float*)d_in[7];
  const float* er0_bb  = (const float*)d_in[8];
  const float* er1_wa  = (const float*)d_in[9];
  const float* er1_ba  = (const float*)d_in[10];
  const float* er1_wb  = (const float*)d_in[11];
  const float* er1_bb  = (const float*)d_in[12];
  const float* enc_wo  = (const float*)d_in[13];
  const float* enc_bo  = (const float*)d_in[14];
  const float* codebook= (const float*)d_in[15];
  const float* dec_w1  = (const float*)d_in[16];
  const float* dec_b1  = (const float*)d_in[17];
  const float* dr0_wa  = (const float*)d_in[18];
  const float* dr0_ba  = (const float*)d_in[19];
  const float* dr0_wb  = (const float*)d_in[20];
  const float* dr0_bb  = (const float*)d_in[21];
  const float* dr1_wa  = (const float*)d_in[22];
  const float* dr1_ba  = (const float*)d_in[23];
  const float* dr1_wb  = (const float*)d_in[24];
  const float* dr1_bb  = (const float*)d_in[25];
  const float* dec_wt  = (const float*)d_in[26];
  const float* dec_bt  = (const float*)d_in[27];
  const float* dec_wo  = (const float*)d_in[28];
  const float* dec_bo  = (const float*)d_in[29];
  float* dout = (float*)d_out;

  // workspace arena
  char* ws = (char*)d_ws;
  size_t off = 0;
  auto alloc = [&](size_t bytes) -> char* {
    char* p = ws + off;
    off = (off + bytes + 255) & ~(size_t)255;
    return p;
  };
  _Float16* W2E  = (_Float16*)alloc((size_t)HC * HC * PATCH * 2);
  _Float16* R0A  = (_Float16*)alloc((size_t)HC * HC * 3 * 2);
  _Float16* R0B  = (_Float16*)alloc((size_t)HC * HC * 2);
  _Float16* R1A  = (_Float16*)alloc((size_t)HC * HC * 3 * 2);
  _Float16* R1B  = (_Float16*)alloc((size_t)HC * HC * 2);
  _Float16* WOE  = (_Float16*)alloc((size_t)EMB * HC * 2);
  _Float16* CBH  = (_Float16*)alloc((size_t)KC * EMB * 2);
  float*    CSQ  = (float*)alloc((size_t)KC * 4);
  _Float16* DW1  = (_Float16*)alloc((size_t)HC * EMB * 3 * 2);
  _Float16* DR0A = (_Float16*)alloc((size_t)HC * HC * 3 * 2);
  _Float16* DR0B = (_Float16*)alloc((size_t)HC * HC * 2);
  _Float16* DR1A = (_Float16*)alloc((size_t)HC * HC * 3 * 2);
  _Float16* DR1B = (_Float16*)alloc((size_t)HC * HC * 2);
  _Float16* WTT  = (_Float16*)alloc((size_t)HC * HC * PATCH * 2);
  float*    SCAL = (float*)alloc((size_t)(KC + 2) * 4);
  _Float16* ACTA = (_Float16*)alloc((size_t)NB * HC * TP * 2);
  _Float16* ACTB = (_Float16*)alloc((size_t)NB * HC * TP * 2);
  float*    ZEF  = (float*)alloc((size_t)NB * EMB * TP * 4);
  _Float16* ZQH  = (_Float16*)alloc((size_t)NB * EMB * TP * 2);
  _Float16* HBIG = (_Float16*)alloc((size_t)NB * HC * TFULL * 2);

  const dim3 blk(256);
  const dim3 gtile(TP / 16, NB);

  k_prep<<<dim3(512), blk, 0, stream>>>(enc_w2, er0_wa, er0_wb, er1_wa, er1_wb,
                                        enc_wo, codebook, dec_w1, dr0_wa, dr0_wb,
                                        dr1_wa, dr1_wb, dec_wt,
                                        W2E, R0A, R0B, R1A, R1B, WOE, CBH, CSQ,
                                        DW1, DR0A, DR0B, DR1A, DR1B, WTT, SCAL);

  // Encoder
  k_enc_front<<<gtile, blk, 0, stream>>>(x, enc_w1, enc_b1, W2E, enc_b2, ACTA);
  k_conv_gemm<<<gtile, blk, 0, stream>>>(ACTA, R0A, er0_ba, nullptr, ACTB, HC, 3, 1, HC, 1, 0);
  k_conv_gemm<<<gtile, blk, 0, stream>>>(ACTB, R0B, er0_bb, ACTA, ACTA, HC, 1, 0, HC, 1, 0);
  k_conv_gemm<<<gtile, blk, 0, stream>>>(ACTA, R1A, er1_ba, nullptr, ACTB, HC, 3, 1, HC, 1, 0);
  k_conv_gemm<<<gtile, blk, 0, stream>>>(ACTB, R1B, er1_bb, ACTA, ACTA, HC, 1, 0, HC, 1, 0);
  k_conv_gemm<<<gtile, blk, 0, stream>>>(ACTA, WOE, enc_bo, nullptr, ZEF, HC, 1, 0, EMB, 1, 1);

  // Vector quantization
  k_quantize<<<dim3(NB * TP / 16), blk, 0, stream>>>(ZEF, CBH, CSQ, codebook, ZQH, dout, SCAL);

  // Decoder
  k_conv_gemm<<<gtile, blk, 0, stream>>>(ZQH, DW1, dec_b1, nullptr, ACTA, EMB, 3, 1, HC, 0, 0);
  k_conv_gemm<<<gtile, blk, 0, stream>>>(ACTA, DR0A, dr0_ba, nullptr, ACTB, HC, 3, 1, HC, 1, 0);
  k_conv_gemm<<<gtile, blk, 0, stream>>>(ACTB, DR0B, dr0_bb, ACTA, ACTA, HC, 1, 0, HC, 1, 0);
  k_conv_gemm<<<gtile, blk, 0, stream>>>(ACTA, DR1A, dr1_ba, nullptr, ACTB, HC, 3, 1, HC, 1, 0);
  k_conv_gemm<<<gtile, blk, 0, stream>>>(ACTB, DR1B, dr1_bb, ACTA, ACTA, HC, 1, 0, HC, 1, 0);
  k_deconv<<<gtile, blk, 0, stream>>>(ACTA, WTT, dec_bt, HBIG);
  k_final<<<dim3(NB * TFULL / 256), blk, 0, stream>>>(HBIG, dec_wo, dec_bo, x, dout, SCAL);
  k_finalize<<<dim3(1), blk, 0, stream>>>(SCAL, dout);
}
